// StackedNetwork_24584392802668
// MI455X (gfx1250) — compile-verified, matrix operand order checked by reading, not run on hardware
//
#include <hip/hip_runtime.h>
#include <hip/hip_bf16.h>

// ---------------------------------------------------------------------------
// Stacked ConvLSTM2D network for MI455X (gfx1250, wave32, WMMA).
// bf16 implicit-GEMM per time step with f32 WMMA accumulation, 2x2 register
// blocking per wave (4 accumulators), halo-row LDS staging via
// GLOBAL_LOAD_ASYNC_TO_LDS_B128, fused LSTM pointwise update + BN,
// ping-pong h state, scalar conv3d+sigmoid epilogue.
// ---------------------------------------------------------------------------

typedef __attribute__((ext_vector_type(16))) __bf16 v16bf;
typedef __attribute__((ext_vector_type(8)))  float  v8f;
typedef __attribute__((ext_vector_type(4)))  int    v4i;

#define BATCH 4
#define TSTEPS 12
#define HH 128
#define WW 128
#define FC 64          // filters per layer
#define NG 256         // 4*FC gate channels

// ---- async global->LDS staging (CDNA5), guarded with sync fallback --------
#if defined(__has_builtin)
#  if __has_builtin(__builtin_amdgcn_global_load_async_to_lds_b128) && \
      __has_builtin(__builtin_amdgcn_s_wait_asynccnt)
#    define USE_ASYNC_LDS 1
#  endif
#endif

__device__ __forceinline__ void stage16(const __bf16* g, __bf16* l) {
#if defined(USE_ASYNC_LDS)
    typedef __attribute__((address_space(1))) v4i* gp_t;   // global int4*
    typedef __attribute__((address_space(3))) v4i* lp_t;   // LDS int4*
    __builtin_amdgcn_global_load_async_to_lds_b128(
        (gp_t)(unsigned long long)(uintptr_t)g,
        (lp_t)(unsigned int)(uintptr_t)l,
        0, 0);
#else
    *(uint4*)l = *(const uint4*)g;
#endif
}
__device__ __forceinline__ void stage_wait() {
#if defined(USE_ASYNC_LDS)
    __builtin_amdgcn_s_wait_asynccnt(0);
#endif
}

__device__ __forceinline__ __bf16 f2bf(float f) {
    unsigned u = __float_as_uint(f);
    unsigned r = (u + 0x7FFFu + ((u >> 16) & 1u)) >> 16;   // round-nearest-even
    unsigned short s = (unsigned short)r;
    __bf16 b; __builtin_memcpy(&b, &s, 2); return b;
}
__device__ __forceinline__ float bf2f(__bf16 b) {
    unsigned short s; __builtin_memcpy(&s, &b, 2);
    return __uint_as_float((unsigned)s << 16);
}
__device__ __forceinline__ float hsig(float x) {
    return fminf(fmaxf(0.2f * x + 0.5f, 0.f), 1.f);
}

// ---------------------------------------------------------------------------
// Weight pack: w [taps][Cin][256] f32 -> out [tap][ck][n(256)][kk(32)] bf16,
// Cin zero-padded to nck*32.
// ---------------------------------------------------------------------------
__global__ void pack_w_kernel(const float* __restrict__ w, __bf16* __restrict__ out,
                              int taps, int Cin, int nck) {
    int total = taps * nck * NG * 32;
    int i = blockIdx.x * 256 + threadIdx.x;
    if (i >= total) return;
    int kk   = i & 31;
    int n    = (i >> 5) & 255;
    int rest = i >> 13;            // tap*nck + ck
    int ck   = rest % nck;
    int tap  = rest / nck;
    int ci   = ck * 32 + kk;
    float v = (ci < Cin) ? w[((size_t)tap * Cin + ci) * NG + n] : 0.f;
    out[i] = f2bf(v);
}

// ---------------------------------------------------------------------------
// Input pack: in [B,T,H,W,1] f32 -> out [T][B][H][W][32] bf16 (channel 0 real).
// ---------------------------------------------------------------------------
__global__ void pack_x0_kernel(const float* __restrict__ in, __bf16* __restrict__ out) {
    long long i = (long long)blockIdx.x * 256 + threadIdx.x;
    const long long total = (long long)TSTEPS * BATCH * HH * WW * 32;
    if (i >= total) return;
    int c = (int)(i & 31);
    long long p = i >> 5;                 // ((t*B+b)*H+y)*W+x
    int x = (int)(p & (WW - 1)); p >>= 7;
    int y = (int)(p & (HH - 1)); p >>= 7;
    int b = (int)(p & (BATCH - 1)); p >>= 2;
    int t = (int)p;
    float v = 0.f;
    if (c == 0)
        v = in[(((long long)b * TSTEPS + t) * HH + y) * WW + x];
    out[i] = f2bf(v);
}

// ---------------------------------------------------------------------------
// Fused ConvLSTM time step: gates via WMMA implicit GEMM + pointwise update.
// Block: 128 threads = 4 waves, wave = gate. Each wave: 2M x 2N accumulators.
// Block tile: 32 pixels (one row segment) x 32 channels, all 4 gates.
//   grid.x = B*H*W/32 = 2048,  grid.y = FC/32 = 2.
// ---------------------------------------------------------------------------
__global__ void __launch_bounds__(128)
convlstm_step_kernel(const __bf16* __restrict__ xslice,  // [B,H,W,Cx]
                     const __bf16* __restrict__ hprev,   // [B,H,W,FC]
                     const __bf16* __restrict__ wxp,     // packed [tap][ckx][256][32]
                     const __bf16* __restrict__ whp,     // packed [tap][2][256][32]
                     const float*  __restrict__ bias,    // [256]
                     const float*  __restrict__ gamma,   // [FC] or null
                     const float*  __restrict__ beta,
                     const float*  __restrict__ mean,
                     const float*  __restrict__ var,
                     float*        __restrict__ cstate,  // [B,H,W,FC] f32
                     __bf16*       __restrict__ hnext,   // [B,H,W,FC]
                     __bf16*       __restrict__ yout,    // [B,H,W,FC] (time slice)
                     int ksize, int Cx, int t_is_zero) {
    __shared__ __align__(32) __bf16 As[36 * 64];          // halo row: (32+ks-1) x C
    __shared__ float zbuf[4][32][32];                     // gate, pixel, channel

    const int tid  = threadIdx.x;
    const int lane = tid & 31;
    const int gate = tid >> 5;                            // wave id == gate
    const int c32  = blockIdx.y;                          // 32-channel group
    const int ptile = blockIdx.x;

    const int xbase = (ptile * 32) & (WW - 1);
    const int rowid = (ptile * 32) / WW;                  // b*H + y
    const int yrow  = rowid & (HH - 1);
    const int bidx  = rowid / HH;

    v8f acc[2][2] = {};

    auto gemm_src = [&](const __bf16* __restrict__ src, const __bf16* __restrict__ wp,
                        int ks, int C) {
        const int r    = ks >> 1;
        const int halo = 32 + ks - 1;
        const int nck  = C >> 5;
        const int nch8 = C >> 3;
        for (int ky = 0; ky < ks; ++ky) {
            const int sy = yrow + ky - r;
            __syncthreads();
            // Stage one halo row: halo pixels x C bf16 channels (16B chunks).
            for (int i = tid; i < halo * nch8; i += 128) {
                const int pix = i / nch8;                 // global x = xbase+pix-r
                const int ch8 = i % nch8;
                const int sx  = xbase + pix - r;
                __bf16* dst = &As[pix * C + ch8 * 8];
                if (sy >= 0 && sy < HH && sx >= 0 && sx < WW) {
                    const __bf16* gsrc = src +
                        ((((size_t)bidx * HH + sy) * WW + sx) * C + ch8 * 8);
                    stage16(gsrc, dst);
                } else {
                    *(uint4*)dst = make_uint4(0u, 0u, 0u, 0u);
                }
            }
            stage_wait();
            __syncthreads();
            const int khalf = lane >> 4;
            const int l15   = lane & 15;
            for (int kx = 0; kx < ks; ++kx) {
                const int tap = ky * ks + kx;
                for (int ck = 0; ck < nck; ++ck) {
                    // Two A fragments (pixel halves), reused across 2 N tiles.
                    v16bf a0 = *(const v16bf*)(&As[(l15 + kx) * C + ck * 32 + khalf * 16]);
                    v16bf a1 = *(const v16bf*)(&As[(l15 + 16 + kx) * C + ck * 32 + khalf * 16]);
                    // Two B fragments (N subtiles), reused across 2 M tiles.
                    const __bf16* bbase = wp +
                        ((((size_t)(tap * nck + ck)) * NG + gate * 64 + c32 * 32 + l15) * 32
                         + khalf * 16);
                    __builtin_prefetch(bbase + NG * 32, 0, 1);
                    v16bf b0 = *(const v16bf*)bbase;
                    v16bf b1 = *(const v16bf*)(bbase + 16 * 32);
                    acc[0][0] = __builtin_amdgcn_wmma_f32_16x16x32_bf16(
                        false, a0, false, b0, (short)0, acc[0][0], false, false);
                    acc[0][1] = __builtin_amdgcn_wmma_f32_16x16x32_bf16(
                        false, a0, false, b1, (short)0, acc[0][1], false, false);
                    acc[1][0] = __builtin_amdgcn_wmma_f32_16x16x32_bf16(
                        false, a1, false, b0, (short)0, acc[1][0], false, false);
                    acc[1][1] = __builtin_amdgcn_wmma_f32_16x16x32_bf16(
                        false, a1, false, b1, (short)0, acc[1][1], false, false);
                }
            }
        }
    };

    gemm_src(xslice, wxp, ksize, Cx);
    if (!t_is_zero)
        gemm_src(hprev, whp, ksize, FC);

    // Bias + scatter z into LDS (C/D layout: N = lane%16, M = vgpr + 8*(lane/16)).
    {
        const int nloc  = lane & 15;
        const int mhalf = lane >> 4;
#pragma unroll
        for (int n = 0; n < 2; ++n) {
            const float bb = bias[gate * 64 + c32 * 32 + n * 16 + nloc];
#pragma unroll
            for (int m = 0; m < 2; ++m)
#pragma unroll
                for (int rr = 0; rr < 8; ++rr)
                    zbuf[gate][m * 16 + mhalf * 8 + rr][n * 16 + nloc] =
                        acc[m][n][rr] + bb;
        }
    }
    __syncthreads();

    // Pointwise LSTM update + BN: 32 pixels x 32 channels, 8 per thread.
    for (int i = tid; i < 1024; i += 128) {
        const int pix = i >> 5;
        const int ch  = i & 31;
        const float zi = zbuf[0][pix][ch];
        const float zf = zbuf[1][pix][ch];
        const float zc = zbuf[2][pix][ch];
        const float zo = zbuf[3][pix][ch];
        const size_t gidx = (((size_t)bidx * HH + yrow) * WW + (xbase + pix)) * FC
                            + c32 * 32 + ch;
        const float cprev = t_is_zero ? 0.f : cstate[gidx];
        const float ig = hsig(zi);
        const float fg = hsig(zf);
        const float cn = fg * cprev + ig * fmaxf(zc, 0.f);
        const float hn = hsig(zo) * fmaxf(cn, 0.f);
        cstate[gidx] = cn;
        hnext[gidx]  = f2bf(hn);
        float yv = hn;
        if (gamma != nullptr) {
            const int cc = c32 * 32 + ch;
            yv = gamma[cc] * (hn - mean[cc]) * rsqrtf(var[cc] + 1e-3f) + beta[cc];
        }
        yout[gidx] = f2bf(yv);
    }
}

// ---------------------------------------------------------------------------
// Final 3x3x3 conv (64 -> 1) + sigmoid.  y2: [T][B][H][W][64] bf16.
// out: [B][T][H][W] f32.
// ---------------------------------------------------------------------------
__global__ void conv3d_sigmoid_kernel(const __bf16* __restrict__ y2,
                                      const float* __restrict__ w,   // [3][3][3][64]
                                      const float* __restrict__ b3,
                                      float* __restrict__ out) {
    long long i = (long long)blockIdx.x * 256 + threadIdx.x;
    const long long total = (long long)BATCH * TSTEPS * HH * WW;
    if (i >= total) return;
    const int x  = (int)(i & (WW - 1));
    const int y  = (int)((i >> 7) & (HH - 1));
    const int bt = (int)(i >> 14);
    const int t  = bt % TSTEPS;
    const int b  = bt / TSTEPS;
    float acc = b3[0];
    for (int dt = -1; dt <= 1; ++dt) {
        const int st = t + dt;
        if (st < 0 || st >= TSTEPS) continue;
        for (int dy = -1; dy <= 1; ++dy) {
            const int sy = y + dy;
            if (sy < 0 || sy >= HH) continue;
            for (int dx = -1; dx <= 1; ++dx) {
                const int sx = x + dx;
                if (sx < 0 || sx >= WW) continue;
                const __bf16* src = y2 +
                    ((((size_t)st * BATCH + b) * HH + sy) * WW + sx) * FC;
                const float* ww = w + (((dt + 1) * 3 + (dy + 1)) * 3 + (dx + 1)) * FC;
                float s = 0.f;
#pragma unroll 8
                for (int c = 0; c < FC; ++c) s += bf2f(src[c]) * ww[c];
                acc += s;
            }
        }
    }
    out[i] = 1.f / (1.f + expf(-acc));
}

// ---------------------------------------------------------------------------
// Host launcher
// ---------------------------------------------------------------------------
extern "C" void kernel_launch(void* const* d_in, const int* in_sizes, int n_in,
                              void* d_out, int out_size, void* d_ws, size_t ws_size,
                              hipStream_t stream) {
    (void)in_sizes; (void)n_in; (void)out_size; (void)ws_size;
    const float* inT  = (const float*)d_in[0];
    const float* wx0  = (const float*)d_in[1];
    const float* wh0  = (const float*)d_in[2];
    const float* b0   = (const float*)d_in[3];
    const float* g0   = (const float*)d_in[4];
    const float* be0  = (const float*)d_in[5];
    const float* m0   = (const float*)d_in[6];
    const float* v0   = (const float*)d_in[7];
    const float* wx1  = (const float*)d_in[8];
    const float* wh1  = (const float*)d_in[9];
    const float* b1   = (const float*)d_in[10];
    const float* g1   = (const float*)d_in[11];
    const float* be1  = (const float*)d_in[12];
    const float* m1   = (const float*)d_in[13];
    const float* v1   = (const float*)d_in[14];
    const float* wx2  = (const float*)d_in[15];
    const float* wh2  = (const float*)d_in[16];
    const float* b2   = (const float*)d_in[17];
    const float* w3d  = (const float*)d_in[18];
    const float* b3d  = (const float*)d_in[19];

    // ---- workspace carve (256B aligned regions) ----
    char*  ws  = (char*)d_ws;
    size_t off = 0;
    auto carve = [&](size_t bytes) -> void* {
        void* p = ws + off;
        off = (off + bytes + 255) & ~(size_t)255;
        return p;
    };
    const size_t seq64  = (size_t)TSTEPS * BATCH * HH * WW * FC * sizeof(__bf16);
    const size_t state2 = (size_t)BATCH * HH * WW * FC * sizeof(__bf16);
    const size_t state4 = (size_t)BATCH * HH * WW * FC * sizeof(float);

    __bf16* wxp0 = (__bf16*)carve((size_t)25 * 1 * NG * 32 * 2);
    __bf16* whp0 = (__bf16*)carve((size_t)25 * 2 * NG * 32 * 2);
    __bf16* wxp1 = (__bf16*)carve((size_t)9  * 2 * NG * 32 * 2);
    __bf16* whp1 = (__bf16*)carve((size_t)9  * 2 * NG * 32 * 2);
    __bf16* wxp2 = (__bf16*)carve((size_t)9  * 2 * NG * 32 * 2);
    __bf16* whp2 = (__bf16*)carve((size_t)9  * 2 * NG * 32 * 2);
    __bf16* y0   = (__bf16*)carve(seq64);
    __bf16* y1   = (__bf16*)carve(seq64);   // also hosts x0pack before layer-1 runs
    __bf16* y2   = (__bf16*)carve(seq64);
    __bf16* hA   = (__bf16*)carve(state2);
    __bf16* hB   = (__bf16*)carve(state2);
    float*  cst  = (float*)carve(state4);
    __bf16* x0p  = y1;   // alias: x0pack is dead before y1 is written

    // ---- weight / input packing ----
    auto pw = [&](const float* w, __bf16* dst, int taps, int Cin, int nck) {
        int total = taps * nck * NG * 32;
        pack_w_kernel<<<(total + 255) / 256, 256, 0, stream>>>(w, dst, taps, Cin, nck);
    };
    pw(wx0, wxp0, 25, 1, 1);
    pw(wh0, whp0, 25, FC, 2);
    pw(wx1, wxp1, 9, FC, 2);
    pw(wh1, whp1, 9, FC, 2);
    pw(wx2, wxp2, 9, FC, 2);
    pw(wh2, whp2, 9, FC, 2);
    {
        long long total = (long long)TSTEPS * BATCH * HH * WW * 32;
        pack_x0_kernel<<<(unsigned)((total + 255) / 256), 256, 0, stream>>>(inT, x0p);
    }

    // ---- layer descriptors ----
    struct Layer {
        const __bf16 *xseq, *wxp, *whp;
        const float *bias, *gamma, *beta, *mean, *var;
        __bf16* yseq;
        int ks, Cx;
    };
    Layer L[3] = {
        { x0p, wxp0, whp0, b0, g0, be0, m0, v0, y0, 5, 32 },
        { y0,  wxp1, whp1, b1, g1, be1, m1, v1, y1, 3, FC },
        { y1,  wxp2, whp2, b2, nullptr, nullptr, nullptr, nullptr, y2, 3, FC },
    };

    const dim3 grid(BATCH * HH * WW / 32, FC / 32);
    for (int l = 0; l < 3; ++l) {
        const size_t xSlice = (size_t)BATCH * HH * WW * L[l].Cx;
        const size_t ySlice = (size_t)BATCH * HH * WW * FC;
        for (int t = 0; t < TSTEPS; ++t) {
            const __bf16* hin  = (t & 1) ? hB : hA;
            __bf16*       hout = (t & 1) ? hA : hB;
            convlstm_step_kernel<<<grid, 128, 0, stream>>>(
                L[l].xseq + (size_t)t * xSlice, hin,
                L[l].wxp, L[l].whp, L[l].bias,
                L[l].gamma, L[l].beta, L[l].mean, L[l].var,
                cst, hout, L[l].yseq + (size_t)t * ySlice,
                L[l].ks, L[l].Cx, t == 0 ? 1 : 0);
        }
    }

    // ---- final conv3d + sigmoid ----
    {
        long long total = (long long)BATCH * TSTEPS * HH * WW;
        conv3d_sigmoid_kernel<<<(unsigned)((total + 255) / 256), 256, 0, stream>>>(
            y2, w3d, b3d, (float*)d_out);
    }
}
